// baseGCN_33775622815788
// MI455X (gfx1250) — matrix operand order repaired and checked by reading, not compiled, not run on hardware
//
#include <hip/hip_runtime.h>
#include <hip/hip_bf16.h>

typedef __attribute__((ext_vector_type(16))) _Float16 v16h;
typedef __attribute__((ext_vector_type(8)))  float    v8f;

union FragH { uint4 q[2]; v16h h; };
union HalfPack { _Float16 h[2]; unsigned int u; };

#define D_FEAT 256

// ---------------------------------------------------------------------------
// Utility kernels
// ---------------------------------------------------------------------------
__global__ void zero_f32_kernel(float* __restrict__ p, long long n) {
    long long t = (long long)blockIdx.x * blockDim.x + threadIdx.x;
    if (t < n) p[t] = 0.0f;
}

__global__ void f32_to_f16_kernel(const float* __restrict__ in, _Float16* __restrict__ out, long long n) {
    long long t = (long long)blockIdx.x * blockDim.x + threadIdx.x;
    if (t < n) out[t] = (_Float16)in[t];
}

// ---------------------------------------------------------------------------
// Degree / normalization
// ---------------------------------------------------------------------------
__global__ void deg_count_kernel(const int* __restrict__ dst, float* __restrict__ deg, int E) {
    int e = blockIdx.x * blockDim.x + threadIdx.x;
    if (e < E) atomicAdd(&deg[dst[e]], 1.0f);
}

__global__ void dinv_kernel(float* __restrict__ deg, int N) {
    int n = blockIdx.x * blockDim.x + threadIdx.x;
    if (n < N) deg[n] = rsqrtf(1.0f + deg[n]);
}

__global__ void coef_kernel(const int* __restrict__ src, const int* __restrict__ dst,
                            const float* __restrict__ dinv, float* __restrict__ coef, int E) {
    int e = blockIdx.x * blockDim.x + threadIdx.x;
    if (e < E) coef[e] = dinv[src[e]] * dinv[dst[e]];
}

// ---------------------------------------------------------------------------
// Pack W [K,256] f32 row-major into the WMMA B-fragment layout (f16 pairs).
// Index: (((ks*16 + coltile)*32 + lane)*8 + j) holds {W[k,col], W[k+1,col]}
//   col = coltile*16 + (lane&15),  k = ks*32 + (lane>=16 ? 8 : 0) + 2*j + (j>=4 ? 8 : 0)
// ---------------------------------------------------------------------------
__global__ void pack_w_kernel(const float* __restrict__ W, unsigned int* __restrict__ Wp, int K) {
    int t = blockIdx.x * blockDim.x + threadIdx.x;
    int total = (K >> 5) * 16 * 32 * 8;
    if (t >= total) return;
    int j    = t & 7;
    int lane = (t >> 3) & 31;
    int ct   = (t >> 8) & 15;
    int ks   = t >> 12;
    int col  = ct * 16 + (lane & 15);
    int k    = ks * 32 + ((lane & 16) ? 8 : 0) + 2 * j + ((j >= 4) ? 8 : 0);
    HalfPack hp;
    hp.h[0] = (_Float16)W[(size_t)k * D_FEAT + col];
    hp.h[1] = (_Float16)W[(size_t)(k + 1) * D_FEAT + col];
    Wp[t] = hp.u;
}

// ---------------------------------------------------------------------------
// WMMA GEMM: C[N,256] = A[N,K](f16 row-major) * W(packed f16)
// block = 128 threads (4 waves); wave tile = 32 rows x 64 cols
// grid = (256/64, ceil(N/128))
// ---------------------------------------------------------------------------
__global__ void gemm_wmma_kernel(const _Float16* __restrict__ A,
                                 const unsigned int* __restrict__ Bp,
                                 float* __restrict__ C, int N, int K) {
    const int lane    = threadIdx.x & 31;
    const int wave    = threadIdx.x >> 5;
    const int rowBase = blockIdx.y * 128 + wave * 32;
    const int colBase = blockIdx.x * 64;

    v8f acc[2][4] = {};

    int mr0 = rowBase + (lane & 15);
    int mr1 = rowBase + 16 + (lane & 15);
    const int r0 = (mr0 < N) ? mr0 : (N - 1);
    const int r1 = (mr1 < N) ? mr1 : (N - 1);
    const int ksteps = K >> 5;

    for (int ks = 0; ks < ksteps; ++ks) {
        const int kbase = ks * 32 + ((lane & 16) ? 8 : 0);
        const uint4* a0p = reinterpret_cast<const uint4*>(A + (size_t)r0 * K + kbase);
        const uint4* a1p = reinterpret_cast<const uint4*>(A + (size_t)r1 * K + kbase);
        FragH a0, a1;
        a0.q[0] = a0p[0]; a0.q[1] = a0p[2];   // K: kbase..+7, kbase+16..+23
        a1.q[0] = a1p[0]; a1.q[1] = a1p[2];
#pragma unroll
        for (int j = 0; j < 4; ++j) {
            const int ct = blockIdx.x * 4 + j;
            const uint4* bp = reinterpret_cast<const uint4*>(
                Bp + (((size_t)ks * 16 + ct) * 32 + lane) * 8);
            FragH b;
            b.q[0] = bp[0]; b.q[1] = bp[1];
            acc[0][j] = __builtin_amdgcn_wmma_f32_16x16x32_f16(
                false, a0.h, false, b.h, (short)0, acc[0][j], false, false);
            acc[1][j] = __builtin_amdgcn_wmma_f32_16x16x32_f16(
                false, a1.h, false, b.h, (short)0, acc[1][j], false, false);
        }
    }

    const int colLane = colBase + (lane & 15);
    const int rOff    = (lane >> 4) * 8;  // lanes 16-31 hold M = 8..15
#pragma unroll
    for (int i = 0; i < 2; ++i) {
        const int rowT = rowBase + i * 16 + rOff;
#pragma unroll
        for (int j = 0; j < 4; ++j) {
            const int col = colLane + j * 16;
#pragma unroll
            for (int r = 0; r < 8; ++r) {
                const int row = rowT + r;
                if (row < N) C[(size_t)row * D_FEAT + col] = acc[i][j][r];
            }
        }
    }
}

// ---------------------------------------------------------------------------
// Aggregation: agg = h*dinv^2 + bias (self loop), then edge scatter-adds
// ---------------------------------------------------------------------------
__global__ void init_agg_kernel(const float* __restrict__ h, const float* __restrict__ dinv,
                                const float* __restrict__ bias, float* __restrict__ agg,
                                long long total) {
    long long t = (long long)blockIdx.x * blockDim.x + threadIdx.x;
    if (t >= total) return;
    int n = (int)(t >> 8);
    int d = (int)(t & 255);
    float dv = dinv[n];
    agg[t] = h[t] * dv * dv + bias[d];
}

__global__ void edge_agg_kernel(const int* __restrict__ src, const int* __restrict__ dst,
                                const float* __restrict__ coef, const float* __restrict__ h,
                                float* __restrict__ agg, int E) {
    long long t = (long long)blockIdx.x * blockDim.x + threadIdx.x;
    int e  = (int)(t >> 6);
    int c4 = (int)(t & 63);
    if (e >= E) return;
    int s = src[e], d = dst[e];
    float cf = coef[e];
    float4 hv = reinterpret_cast<const float4*>(h + (size_t)s * D_FEAT)[c4];
    float* out = agg + (size_t)d * D_FEAT + c4 * 4;
    atomicAdd(out + 0, hv.x * cf);
    atomicAdd(out + 1, hv.y * cf);
    atomicAdd(out + 2, hv.z * cf);
    atomicAdd(out + 3, hv.w * cf);
}

// ---------------------------------------------------------------------------
// BatchNorm (training mode, biased variance)
// stats layout: [0:256) sum, [256:512) sumsq, [512:768) mu, [768:1024) rsig
// ---------------------------------------------------------------------------
#define BN_ROWS 256
__global__ void bn_stats_kernel(const float* __restrict__ agg, float* __restrict__ stats, int N) {
    const int d  = threadIdx.x;
    int r0 = blockIdx.x * BN_ROWS;
    int r1 = r0 + BN_ROWS; if (r1 > N) r1 = N;
    float s = 0.0f, s2 = 0.0f;
    for (int r = r0; r < r1; ++r) {
        float v = agg[(size_t)r * D_FEAT + d];
        s += v; s2 += v * v;
    }
    atomicAdd(&stats[d], s);
    atomicAdd(&stats[D_FEAT + d], s2);
}

__global__ void bn_finalize_kernel(float* __restrict__ stats, int N) {
    int d = threadIdx.x;
    float inv_n = 1.0f / (float)N;
    float mu  = stats[d] * inv_n;
    float var = stats[D_FEAT + d] * inv_n - mu * mu;
    stats[2 * D_FEAT + d] = mu;
    stats[3 * D_FEAT + d] = rsqrtf(var + 1e-5f);
}

__global__ void bn_tanh_kernel(const float* __restrict__ agg, const float* __restrict__ stats,
                               const float* __restrict__ g, const float* __restrict__ be,
                               _Float16* __restrict__ h16, float* __restrict__ hf,
                               long long total) {
    long long t = (long long)blockIdx.x * blockDim.x + threadIdx.x;
    if (t >= total) return;
    int d = (int)(t & 255);
    float v = (agg[t] - stats[2 * D_FEAT + d]) * stats[3 * D_FEAT + d] * g[d] + be[d];
    float y = tanhf(v);
    hf[t]  = y;
    h16[t] = (_Float16)y;
}

// ---------------------------------------------------------------------------
// Pooling: order-preserving uint keys for float atomicMax
// ---------------------------------------------------------------------------
__device__ __forceinline__ unsigned int fkey(float f) {
    unsigned int u = __float_as_uint(f);
    return (u & 0x80000000u) ? ~u : (u | 0x80000000u);
}
__device__ __forceinline__ float funkey(unsigned int k) {
    unsigned int u = (k & 0x80000000u) ? (k & 0x7FFFFFFFu) : ~k;
    return __uint_as_float(u);
}

__global__ void pool_init_kernel(unsigned int* __restrict__ hmaxbits, float* __restrict__ hsum,
                                 float* __restrict__ cnt, int B) {
    int t = blockIdx.x * blockDim.x + threadIdx.x;
    int total = B * D_FEAT;
    if (t < total) { hmaxbits[t] = 0x007FFFFFu; hsum[t] = 0.0f; }  // key(-inf)
    if (t < B) cnt[t] = 0.0f;
}

__global__ void pool_kernel(const float* __restrict__ hf, const int* __restrict__ batch,
                            unsigned int* __restrict__ hmaxbits, float* __restrict__ hsum,
                            float* __restrict__ cnt, long long total) {
    long long t = (long long)blockIdx.x * blockDim.x + threadIdx.x;
    if (t >= total) return;
    int n = (int)(t >> 8);
    int d = (int)(t & 255);
    int b = batch[n];
    float v = hf[t];
    atomicMax(&hmaxbits[(size_t)b * D_FEAT + d], fkey(v));
    atomicAdd(&hsum[(size_t)b * D_FEAT + d], v);
    if (d == 0) atomicAdd(&cnt[b], 1.0f);
}

__global__ void pool_finalize_kernel(const unsigned int* __restrict__ hmaxbits,
                                     const float* __restrict__ hsum, const float* __restrict__ cnt,
                                     float* __restrict__ hidden, int B) {
    int t = blockIdx.x * blockDim.x + threadIdx.x;
    if (t >= B * D_FEAT) return;
    int b = t >> 8, d = t & 255;
    hidden[(size_t)b * 512 + d]       = funkey(hmaxbits[t]);
    hidden[(size_t)b * 512 + 256 + d] = hsum[t] / cnt[b];
}

__global__ void out_kernel(const float* __restrict__ hidden, const float* __restrict__ Wout,
                           const float* __restrict__ bout, float* __restrict__ out, int B) {
    int b = blockIdx.x * blockDim.x + threadIdx.x;
    if (b >= B) return;
    float s = bout[0];
    const float* h = hidden + (size_t)b * 512;
#pragma unroll 8
    for (int k = 0; k < 512; ++k) s += h[k] * Wout[k];
    out[b] = s;
}

// ---------------------------------------------------------------------------
// Host launch
// ---------------------------------------------------------------------------
extern "C" void kernel_launch(void* const* d_in, const int* in_sizes, int n_in,
                              void* d_out, int out_size, void* d_ws, size_t ws_size,
                              hipStream_t stream) {
    const float* x     = (const float*)d_in[0];
    const int*   ei    = (const int*)d_in[1];
    const int*   batch = (const int*)d_in[2];
    const float* Ws[4]  = {(const float*)d_in[4],  (const float*)d_in[6],
                           (const float*)d_in[8],  (const float*)d_in[10]};
    const float* bs[4]  = {(const float*)d_in[5],  (const float*)d_in[7],
                           (const float*)d_in[9],  (const float*)d_in[11]};
    const float* gs[4]  = {(const float*)d_in[12], (const float*)d_in[14],
                           (const float*)d_in[16], (const float*)d_in[18]};
    const float* bes[4] = {(const float*)d_in[13], (const float*)d_in[15],
                           (const float*)d_in[17], (const float*)d_in[19]};
    const float* Wout = (const float*)d_in[20];
    const float* bout = (const float*)d_in[21];

    const int N = in_sizes[0] / 64;       // 50000
    const int E = in_sizes[1] / 2;        // 800000
    const int B = out_size / 513;         // 1000
    const int* srcI = ei;
    const int* dstI = ei + E;

    // ---- workspace carve-up ----
    char* p = (char*)d_ws;
    auto alloc = [&](size_t bytes) {
        void* r = (void*)p;
        p += (bytes + 255) & ~(size_t)255;
        return r;
    };
    _Float16*     h16   = (_Float16*)alloc((size_t)N * D_FEAT * 2);   // layer inputs (layer0: [N,64])
    float*        htmp  = (float*)alloc((size_t)N * D_FEAT * 4);      // GEMM out / hfinal
    float*        agg   = (float*)alloc((size_t)N * D_FEAT * 4);
    float*        dinv  = (float*)alloc((size_t)N * 4);
    float*        coef  = (float*)alloc((size_t)E * 4);
    unsigned int* w16p  = (unsigned int*)alloc((size_t)8 * 16 * 32 * 8 * 4);
    float*        stats = (float*)alloc((size_t)4 * D_FEAT * 4);
    unsigned int* hmaxb = (unsigned int*)alloc((size_t)B * D_FEAT * 4);
    float*        hsum  = (float*)alloc((size_t)B * D_FEAT * 4);
    float*        cnt   = (float*)alloc((size_t)B * 4);

    float* out    = (float*)d_out;
    float* hidden = out + B;

    const long long totND = (long long)N * D_FEAT;
    auto blocks1d = [](long long n, int bs_) { return (unsigned int)((n + bs_ - 1) / bs_); };

    // ---- degree / dinv / coef ----
    zero_f32_kernel<<<blocks1d(N, 256), 256, 0, stream>>>(dinv, N);
    deg_count_kernel<<<blocks1d(E, 256), 256, 0, stream>>>(dstI, dinv, E);
    dinv_kernel<<<blocks1d(N, 256), 256, 0, stream>>>(dinv, N);
    coef_kernel<<<blocks1d(E, 256), 256, 0, stream>>>(srcI, dstI, dinv, coef, E);

    // ---- convert x -> f16 input (stored in h16 region as [N,64]) ----
    f32_to_f16_kernel<<<blocks1d((long long)N * 64, 256), 256, 0, stream>>>(
        x, h16, (long long)N * 64);

    // ---- 4 GCN layers ----
    for (int l = 0; l < 4; ++l) {
        const int K = (l == 0) ? 64 : D_FEAT;
        const int packTot = (K >> 5) * 16 * 32 * 8;
        pack_w_kernel<<<blocks1d(packTot, 256), 256, 0, stream>>>(Ws[l], w16p, K);

        gemm_wmma_kernel<<<dim3(4, (N + 127) / 128), dim3(128), 0, stream>>>(
            h16, w16p, htmp, N, K);

        init_agg_kernel<<<blocks1d(totND, 256), 256, 0, stream>>>(
            htmp, dinv, bs[l], agg, totND);
        edge_agg_kernel<<<blocks1d((long long)E * 64, 256), 256, 0, stream>>>(
            srcI, dstI, coef, htmp, agg, E);

        zero_f32_kernel<<<2, 256, 0, stream>>>(stats, 2 * D_FEAT);
        bn_stats_kernel<<<blocks1d(N, BN_ROWS), D_FEAT, 0, stream>>>(agg, stats, N);
        bn_finalize_kernel<<<1, D_FEAT, 0, stream>>>(stats, N);

        bn_tanh_kernel<<<blocks1d(totND, 256), 256, 0, stream>>>(
            agg, stats, gs[l], bes[l], h16, htmp, totND);
    }

    // ---- pooling + head (hfinal == htmp) ----
    pool_init_kernel<<<blocks1d((long long)B * D_FEAT, 256), 256, 0, stream>>>(
        hmaxb, hsum, cnt, B);
    pool_kernel<<<blocks1d(totND, 256), 256, 0, stream>>>(
        htmp, batch, hmaxb, hsum, cnt, totND);
    pool_finalize_kernel<<<blocks1d((long long)B * D_FEAT, 256), 256, 0, stream>>>(
        hmaxb, hsum, cnt, hidden, B);
    out_kernel<<<blocks1d(B, 256), 256, 0, stream>>>(hidden, Wout, bout, out, B);

    (void)n_in; (void)ws_size; (void)batch;
}
